// Model_39676907885118
// MI455X (gfx1250) — matrix operand order repaired and checked by reading, not compiled, not run on hardware
//
#include <hip/hip_runtime.h>

// ---- WMMA vector types (gfx1250, wave32) ----
typedef __attribute__((ext_vector_type(16))) _Float16 v16h;
typedef __attribute__((ext_vector_type(8)))  _Float16 v8h;
typedef __attribute__((ext_vector_type(4)))  _Float16 v4h;
typedef __attribute__((ext_vector_type(8)))  float    v8f;

#define B_   2
#define H_   16
#define SQ_  2048
#define SK_  2048
#define D_   128
#define SCALE_ 0.20884964425119185f

#define BQ   128      // query rows per block
#define BK   128      // key rows per K-tile
#define LDH  136      // LDS row stride (halfs); 272B rows keep 16B alignment, skew banks
#define NTHREADS 256  // 8 wave32s

// A-fragment (16x32 f16, MxK): lanes 0-15 hold row M=lane, K = kb+0..7 & kb+16..23;
// lanes 16-31 hold row M=lane-16, K = kb+8..15 & kb+24..31.
__device__ __forceinline__ v16h frag_a(const _Float16* base, int row0, int kb) {
  const int lane = threadIdx.x & 31;
  const int rw = lane & 15, hf = lane >> 4;
  const _Float16* p = base + (row0 + rw) * LDH;
  v8h c0 = *(const v8h*)(p + kb + hf * 8);
  v8h c1 = *(const v8h*)(p + kb + 16 + hf * 8);
  return __builtin_shufflevector(c0, c1, 0,1,2,3,4,5,6,7,8,9,10,11,12,13,14,15);
}

// B-fragment (32x16 f16, KxN): lanes 0-15 hold column N=lane, K = kb+0..7 & kb+16..23;
// lanes 16-31 hold column N=lane-16, K = kb+8..15 & kb+24..31. `base` stores B columns as
// contiguous LDS rows (i.e. the K-major / transposed copy), so both chunks are b128s.
__device__ __forceinline__ v16h frag_b(const _Float16* base, int col0, int kb) {
  const int lane = threadIdx.x & 31;
  const int rw = lane & 15, hf = lane >> 4;
  const _Float16* p = base + (col0 + rw) * LDH + kb + hf * 8;
  v8h c0 = *(const v8h*)(p);
  v8h c1 = *(const v8h*)(p + 16);
  return __builtin_shufflevector(c0, c1, 0,1,2,3,4,5,6,7,8,9,10,11,12,13,14,15);
}

__global__ __launch_bounds__(NTHREADS, 1)
void fattn_wmma_kernel(const float* __restrict__ x1,
                       const float* __restrict__ x2,
                       const float* __restrict__ dmask,
                       float* __restrict__ out) {
  extern __shared__ _Float16 smem[];
  _Float16* Qh  = smem;                 // [BQ][LDH]   Q tile, row-major f16
  _Float16* Krm = Qh  + BQ * LDH;       // [BK][LDH]   K tile rows (contig in d)
  _Float16* Vt  = Krm + BK * LDH;       // [D_][LDH]   V tile transposed (contig in k)
  _Float16* Psc = Vt  + D_ * LDH;       // [8*16][LDH] per-wave P scratch

  const int tid  = threadIdx.x;
  const int wid  = tid >> 5;
  const int lane = tid & 31;
  const int rw   = lane & 15;
  const int hf   = lane >> 4;
  const int bh   = blockIdx.y;
  const int q0   = blockIdx.x * BQ;

  // ---- load Q tile (f32 -> f16) into LDS, coalesced b128 global loads ----
  {
    const float* qsrc = x1 + ((size_t)bh * SQ_ + q0) * D_;
#pragma unroll
    for (int i = 0; i < 16; ++i) {
      int idx = tid + i * NTHREADS;           // 4096 float4s = 128x128 f32
      int row = idx >> 5, c4 = (idx & 31) * 4;
      float4 v = *(const float4*)(qsrc + (size_t)row * D_ + c4);
      v4h h = { (_Float16)v.x, (_Float16)v.y, (_Float16)v.z, (_Float16)v.w };
      *(v4h*)(Qh + row * LDH + c4) = h;
    }
  }

  // ---- per-row online-softmax state (8 rows per lane: rows hf*8 + r) ----
  float m_run[8], l_run[8];
  v8f acc_o[8];
#pragma unroll
  for (int r = 0; r < 8; ++r) { m_run[r] = -__builtin_inff(); l_run[r] = 0.0f; }
#pragma unroll
  for (int dt = 0; dt < 8; ++dt)
#pragma unroll
    for (int r = 0; r < 8; ++r) acc_o[dt][r] = 0.0f;

  const float* kvsrc = x2 + (size_t)bh * SK_ * D_;
  const float* mrow0 = dmask + ((size_t)bh * SQ_ + q0 + wid * 16) * (size_t)SK_;
  _Float16* Pw = Psc + wid * 16 * LDH;

  for (int k0 = 0; k0 < SK_; k0 += BK) {
    // prefetch next K tile (reused by 16 blocks -> keep temporal) and next mask
    // tile (streamed once) into GL2 while computing on the current one
    if (k0 + BK < SK_) {
      __builtin_prefetch(kvsrc + (size_t)(k0 + BK + (tid >> 1)) * D_ + (tid & 1) * 64, 0, 1);
      __builtin_prefetch(mrow0 + (size_t)(lane >> 1) * SK_ + k0 + BK + (lane & 1) * 64, 0, 0);
    }

    __syncthreads();
    // ---- cooperative K/V tile load: f32 -> f16, row-major + transposed copies ----
#pragma unroll
    for (int i = 0; i < 16; ++i) {
      int idx = tid + i * NTHREADS;
      int row = idx >> 5, c = (idx & 31) * 4;
      float4 v = *(const float4*)(kvsrc + (size_t)(k0 + row) * D_ + c);
      _Float16 h0 = (_Float16)v.x, h1 = (_Float16)v.y,
               h2 = (_Float16)v.z, h3 = (_Float16)v.w;
      v4h hp = { h0, h1, h2, h3 };
      *(v4h*)(Krm + row * LDH + c) = hp;      // contig in d  (GEMM1 B-operand)
      Vt[(c + 0) * LDH + row] = h0;           // contig in k  (GEMM2 B-operand)
      Vt[(c + 1) * LDH + row] = h1;
      Vt[(c + 2) * LDH + row] = h2;
      Vt[(c + 3) * LDH + row] = h3;
    }
    __syncthreads();

    // ---- GEMM1: S(16 x 128) = Q_wave (16 x 128) * K_tile^T ----
    v8f acc_s[8];
#pragma unroll
    for (int nt = 0; nt < 8; ++nt)
#pragma unroll
      for (int r = 0; r < 8; ++r) acc_s[nt][r] = 0.0f;
#pragma unroll
    for (int kb = 0; kb < D_; kb += 32) {
      v16h a = frag_a(Qh, wid * 16, kb);
#pragma unroll
      for (int nt = 0; nt < 8; ++nt) {
        v16h b = frag_b(Krm, nt * 16, kb);
        acc_s[nt] = __builtin_amdgcn_wmma_f32_16x16x32_f16(
            false, a, false, b, (short)0, acc_s[nt], false, false);
      }
    }

    // ---- online softmax (C-layout: lane rw = column, VGPR r = row hf*8+r) ----
    float rowmax[8];
#pragma unroll
    for (int r = 0; r < 8; ++r) rowmax[r] = -__builtin_inff();
#pragma unroll
    for (int nt = 0; nt < 8; ++nt)
#pragma unroll
      for (int r = 0; r < 8; ++r)
        rowmax[r] = fmaxf(rowmax[r], acc_s[nt][r] * SCALE_);
#pragma unroll
    for (int r = 0; r < 8; ++r)
#pragma unroll
      for (int m = 1; m < 16; m <<= 1)   // stays within the 16-lane half (wave32)
        rowmax[r] = fmaxf(rowmax[r], __shfl_xor(rowmax[r], m, 32));

    float mnew[8], alpha[8], rs[8];
#pragma unroll
    for (int r = 0; r < 8; ++r) {
      mnew[r]  = fmaxf(m_run[r], rowmax[r]);
      alpha[r] = __expf(m_run[r] - mnew[r]);
      rs[r]    = 0.0f;
    }
#pragma unroll
    for (int nt = 0; nt < 8; ++nt)
#pragma unroll
      for (int r = 0; r < 8; ++r) {
        float p = __expf(acc_s[nt][r] * SCALE_ - mnew[r]);
        acc_s[nt][r] = p;                // un-masked p: denominator uses this
        rs[r] += p;
      }
#pragma unroll
    for (int r = 0; r < 8; ++r) {
#pragma unroll
      for (int m = 1; m < 16; m <<= 1)
        rs[r] += __shfl_xor(rs[r], m, 32);
      l_run[r] = l_run[r] * alpha[r] + rs[r];
      m_run[r] = mnew[r];
    }

    // ---- apply dropout mask (post-softmax, per reference) & stage P as f16 ----
    // Mask is a 536 MB stream read exactly once: use non-temporal loads so it
    // does not evict the reused x2 tiles (33.5 MB, fits the 192 MB L2).
#pragma unroll
    for (int nt = 0; nt < 8; ++nt)
#pragma unroll
      for (int r = 0; r < 8; ++r) {
        float mv = __builtin_nontemporal_load(
            mrow0 + (size_t)(hf * 8 + r) * SK_ + k0 + nt * 16 + rw);
        Pw[(hf * 8 + r) * LDH + nt * 16 + rw] = (_Float16)(acc_s[nt][r] * mv);
      }

    // ---- rescale output accumulator by alpha (row-dependent) ----
#pragma unroll
    for (int dt = 0; dt < 8; ++dt)
#pragma unroll
      for (int r = 0; r < 8; ++r) acc_o[dt][r] *= alpha[r];

    // ---- GEMM2: O(16 x 128) += P(16 x 128) * V_tile(128 x 128) ----
#pragma unroll
    for (int kb = 0; kb < BK; kb += 32) {
      v16h a = frag_a(Pw, 0, kb);
#pragma unroll
      for (int dt = 0; dt < 8; ++dt) {
        v16h b = frag_b(Vt, dt * 16, kb);
        acc_o[dt] = __builtin_amdgcn_wmma_f32_16x16x32_f16(
            false, a, false, b, (short)0, acc_o[dt], false, false);
      }
    }
  }

  // ---- epilogue: O = acc / l ; written once, never re-read -> non-temporal ----
  float* obase = out + ((size_t)bh * SQ_ + q0 + wid * 16) * D_;
#pragma unroll
  for (int r = 0; r < 8; ++r) {
    float inv_l = 1.0f / l_run[r];
#pragma unroll
    for (int dt = 0; dt < 8; ++dt)
      __builtin_nontemporal_store(acc_o[dt][r] * inv_l,
          obase + (size_t)(hf * 8 + r) * D_ + dt * 16 + rw);
  }
}

extern "C" void kernel_launch(void* const* d_in, const int* in_sizes, int n_in,
                              void* d_out, int out_size, void* d_ws, size_t ws_size,
                              hipStream_t stream) {
  (void)in_sizes; (void)n_in; (void)d_ws; (void)ws_size; (void)out_size;
  const float* x1    = (const float*)d_in[0];
  const float* x2    = (const float*)d_in[1];
  const float* dmask = (const float*)d_in[2];
  float* out = (float*)d_out;

  dim3 grid(SQ_ / BQ, B_ * H_, 1);     // (16, 32)
  dim3 block(NTHREADS, 1, 1);          // 8 wave32s
  size_t smem = (size_t)(BQ + BK + D_ + 8 * 16) * LDH * sizeof(_Float16); // 139,264 B
  fattn_wmma_kernel<<<grid, block, smem, stream>>>(x1, x2, dmask, out);
}
